// BidirectionalGlobalCluster_68487548502833
// MI455X (gfx1250) — matrix operand order repaired and checked by reading, not compiled, not run on hardware
//
#include <hip/hip_runtime.h>
#include <math.h>

// ---------------- problem constants ----------------
#define CC    256        // channels
#define HH    160
#define WW    160
#define RR    4
#define SH    40
#define SW    40
#define SCD   32         // per-head depth
#define SS    64         // anchors per head
#define LL    1600       // sh*sw
#define PIX   25600      // h*w
#define INV_SCALE 0.17677669529663687f   // 1/sqrt(32)

typedef __attribute__((ext_vector_type(16))) __bf16          v16bf;
typedef __attribute__((ext_vector_type(8)))  float           v8f;
typedef __attribute__((ext_vector_type(8)))  unsigned short  v8us;
typedef __attribute__((ext_vector_type(4)))  unsigned int    v4ui;
typedef __attribute__((ext_vector_type(8)))  int             v8si;
typedef __attribute__((ext_vector_type(4)))  int             v4si;

union BFrag { v16bf v; v8us w[2]; unsigned short u[16]; };

__device__ __forceinline__ unsigned short f2bf(float f) {
  union { float f; unsigned u; } c; c.f = f;
  unsigned u = c.u + 0x7FFFu + ((c.u >> 16) & 1u);   // round-to-nearest-even
  return (unsigned short)(u >> 16);
}

__device__ __forceinline__ v8f v8zero() {
  v8f r = {0.f,0.f,0.f,0.f,0.f,0.f,0.f,0.f};
  return r;
}

__device__ __forceinline__ v8f wmma_bf16(v16bf a, v16bf b, v8f c) {
  return __builtin_amdgcn_wmma_f32_16x16x32_bf16(false, a, false, b, (short)0, c,
                                                 false, false);
}

// LDS byte offset of a __shared__ object (LDS aperture: addr[31:0] = offset)
__device__ __forceinline__ unsigned lds_off(const void* p) {
  return (unsigned)(unsigned long long)p;
}

// -------- Tensor Data Mover: 2-D tile (bf16) global -> LDS --------
// tile_d0 elements contiguous per line, tile_d1 lines, line stride stride0 (elems).
__device__ __forceinline__ void tdm_load_2d(unsigned lds_addr, const void* gaddr,
                                            unsigned tensor_d0, unsigned tensor_d1,
                                            unsigned tile_d0, unsigned tile_d1,
                                            unsigned long long stride0) {
  unsigned long long ga = (unsigned long long)gaddr;
  v4ui g0;
  g0[0] = 1u;                                             // count=1 (valid user D#)
  g0[1] = lds_addr;                                       // lds_addr [63:32]
  g0[2] = (unsigned)(ga & 0xFFFFFFFFu);                   // global_addr[31:0]
  g0[3] = (unsigned)((ga >> 32) & 0x01FFFFFFu) | (2u << 30);  // addr[56:32] | type=2
  v8si g1;
  g1[0] = (int)(1u << 16);                                // data_size=1 (2 bytes)
  g1[1] = (int)((tensor_d0 & 0xFFFFu) << 16);             // tensor_dim0[15:0] @ [63:48]
  g1[2] = (int)((tensor_d0 >> 16) | ((tensor_d1 & 0xFFFFu) << 16));
  g1[3] = (int)((tensor_d1 >> 16) | (tile_d0 << 16));     // tile_dim0 @ [127:112]
  g1[4] = (int)(tile_d1 & 0xFFFFu);                       // tile_dim1, tile_dim2=0
  g1[5] = (int)(unsigned)(stride0 & 0xFFFFFFFFu);         // dim0 stride [191:160]
  g1[6] = (int)(unsigned)((stride0 >> 32) & 0xFFFFu);     // stride[47:32]
  g1[7] = 0;
  v4si z4 = {0, 0, 0, 0};
  v8si z8 = {0, 0, 0, 0, 0, 0, 0, 0};
  __builtin_amdgcn_tensor_load_to_lds(g0, g1, z4, z4, z8, 0);
}

// ---------------- 1. fused depthwise conv 4x4/s4 + maxpool 4x4 ----------------
__global__ void downsample_k(const float* __restrict__ x, const float* __restrict__ wc,
                             const float* __restrict__ bc, float* __restrict__ dq,
                             float* __restrict__ dv) {
  int idx = blockIdx.x * blockDim.x + threadIdx.x;   // 6400*256
  int c = idx & 255;
  int pos = idx >> 8;
  int n = pos / LL; int rem = pos - n * LL;
  int oh = rem / SW, ow = rem - oh * SW;
  const float* xp = x + (((size_t)(n * CC + c) * HH + oh * RR) * WW + ow * RR);
  float acc = bc[c];
  float mx = -1e30f;
#pragma unroll
  for (int i = 0; i < 4; ++i)
#pragma unroll
    for (int j = 0; j < 4; ++j) {
      float v = xp[i * WW + j];
      acc += v * wc[c * 16 + i * 4 + j];
      mx = fmaxf(mx, v);
    }
  dq[(size_t)pos * CC + c] = acc;
  dv[(size_t)pos * CC + c] = mx;
}

// ---------------- 2. LayerNorm over 256 channels -> bf16 ----------------
__global__ void ln_k(const float* __restrict__ in, const float* __restrict__ w,
                     const float* __restrict__ b, unsigned short* __restrict__ outb) {
  __shared__ float red[256];
  int row = blockIdx.x, c = threadIdx.x;
  float v = in[(size_t)row * CC + c];
  red[c] = v; __syncthreads();
  for (int s = 128; s > 0; s >>= 1) { if (c < s) red[c] += red[c + s]; __syncthreads(); }
  float mu = red[0] * (1.f / CC); __syncthreads();
  float d = v - mu;
  red[c] = d * d; __syncthreads();
  for (int s = 128; s > 0; s >>= 1) { if (c < s) red[c] += red[c + s]; __syncthreads(); }
  float var = red[0] * (1.f / CC);
  outb[(size_t)row * CC + c] = f2bf(d * rsqrtf(var + 1e-5f) * w[c] + b[c]);
}

// ---------------- 2b. weight transpose + bf16 convert: Wt[n][k] = W[k][n] -----
__global__ void wtrans_k(const float* __restrict__ W, unsigned short* __restrict__ Wt) {
  int idx = blockIdx.x * blockDim.x + threadIdx.x;   // 65536
  int n = idx >> 8, k = idx & 255;
  Wt[idx] = f2bf(W[(size_t)k * 256 + n]);
}

// ---------------- 2c. x NCHW f32 -> NHWC bf16 ----------------
__global__ void xconv_k(const float* __restrict__ x, unsigned short* __restrict__ xb) {
  size_t idx = (size_t)blockIdx.x * blockDim.x + threadIdx.x;  // 4*25600*256
  int c = idx & 255;
  size_t r = idx >> 8;
  int n = (int)(r / PIX), pix = (int)(r - (size_t)n * PIX);
  xb[idx] = f2bf(x[((size_t)(n * CC + c)) * PIX + pix]);
}

// ---------------- 3. TDM + WMMA GEMM: C[M,256] = A[M,256] @ Wt^T + bias -------
// Ab : bf16 row-major [M][256]; Wt : bf16 [256 n][256 k] (pre-transposed weight)
// c_off <0 -> C f32 row-major; >=0 -> C = out NCHW (N,512,H,W) channel offset
__global__ void gemm256_k(const unsigned short* __restrict__ Ab,
                          const unsigned short* __restrict__ Wt,
                          const float* __restrict__ bias, float* __restrict__ C,
                          int c_off) {
  __shared__ __align__(16) unsigned short As[64][256];   // 32 KB
  __shared__ __align__(16) unsigned short Bs[32][256];   // 16 KB (Bs[col][k])
  int tile_m = blockIdx.x * 64;
  int tile_n = blockIdx.y * 32;
  int tid = threadIdx.x;                 // 128 threads = 4 waves
  int wave = tid >> 5, lane = tid & 31;
  if (tid == 0) {
    tdm_load_2d(lds_off(&As[0][0]), Ab + (size_t)tile_m * 256, 256, 64, 256, 64, 256);
    tdm_load_2d(lds_off(&Bs[0][0]), Wt + (size_t)tile_n * 256, 256, 32, 256, 32, 256);
    __builtin_amdgcn_s_wait_tensorcnt(0);
  }
  __syncthreads();
  int row = lane & 15;
  int kbA = (lane & 16) ? 16 : 0;
  int kbB = (lane & 16) ? 16 : 0;
  int col = lane & 15;
  v8f acc[2] = {v8zero(), v8zero()};
#pragma unroll
  for (int k0 = 0; k0 < 256; k0 += 32) {
    BFrag a;
    a.w[0] = *(const v8us*)&As[wave * 16 + row][k0 + kbA];
    a.w[1] = *(const v8us*)&As[wave * 16 + row][k0 + kbA + 8];
#pragma unroll
    for (int t = 0; t < 2; ++t) {
      BFrag b;
      b.w[0] = *(const v8us*)&Bs[t * 16 + col][k0 + kbB];
      b.w[1] = *(const v8us*)&Bs[t * 16 + col][k0 + kbB + 8];
      acc[t] = wmma_bf16(a.v, b.v, acc[t]);
    }
  }
  int rbase = tile_m + wave * 16 + ((lane & 16) ? 8 : 0);
#pragma unroll
  for (int t = 0; t < 2; ++t) {
    int ccol = tile_n + t * 16 + col;
    float bv = bias[ccol];
#pragma unroll
    for (int i = 0; i < 8; ++i) {
      int r = rbase + i;
      float v = acc[t][i] + bv;
      if (c_off >= 0) {
        int n = r / PIX, pix = r - n * PIX;
        C[((size_t)(n * 512 + c_off + ccol)) * PIX + pix] = v;
      } else {
        C[(size_t)r * 256 + ccol] = v;
      }
    }
  }
}

// ---------------- 4. anchor 5x5 mean pool ----------------
__global__ void anchor_pool_k(const float* __restrict__ src, float* __restrict__ dst) {
  int idx = blockIdx.x * blockDim.x + threadIdx.x;   // 4*64*256
  int c = idx & 255;
  int s = (idx >> 8) & 63;
  int n = idx >> 14;
  int ai = s >> 3, aj = s & 7;
  float sum = 0.f;
#pragma unroll
  for (int i = 0; i < 5; ++i)
    for (int j = 0; j < 5; ++j)
      sum += src[((size_t)(n * LL + (ai * 5 + i) * SW + (aj * 5 + j))) * CC + c];
  dst[idx] = sum * (1.f / 25.f);
}

// ---------------- 5. bidirectional flash attention (WMMA) ----------------
__global__ void attention_k(const float* __restrict__ dqf, const float* __restrict__ dvf,
                            unsigned short* __restrict__ attm) {
  __shared__ __align__(16) unsigned short Ks[32][32];    // [key][dim]
  __shared__ __align__(16) unsigned short VsT[32][32];   // [dim][key]
  __shared__ __align__(16) unsigned short Ps[16][32];
  int qt = blockIdx.x, mh = blockIdx.y;
  int lane = threadIdx.x;
  int batch = mh >> 3, fc = mh & 7;
  int peer = mh ^ 16;
  int pb = peer >> 3, pfc = peer & 7;
  BFrag qa;
  {
    int row = lane & 15, kb = (lane & 16) ? 16 : 0;
    const float* qr = dqf + ((size_t)(batch * LL + qt * 16 + row)) * CC + fc * SCD;
#pragma unroll
    for (int h = 0; h < 16; ++h) {
      int k = kb + h;
      qa.u[h] = f2bf(qr[k] * INV_SCALE);
    }
  }
  float rm[8], rs[8];
  v8f o0 = v8zero(), o1 = v8zero();
#pragma unroll
  for (int i = 0; i < 8; ++i) { rm[i] = -1e30f; rs[i] = 0.f; }

  for (int kt = 0; kt < LL / 32; ++kt) {
    int key = kt * 32 + lane;
    const float* kr = dqf + ((size_t)(pb * LL + key)) * CC + pfc * SCD;
    const float* vr = dvf + ((size_t)(pb * LL + key)) * CC + pfc * SCD;
#pragma unroll
    for (int g = 0; g < 4; ++g) {          // K row: 4 x b128 stores
      v8us t;
#pragma unroll
      for (int j = 0; j < 8; ++j) t[j] = f2bf(kr[g * 8 + j]);
      *(v8us*)&Ks[lane][g * 8] = t;
    }
#pragma unroll
    for (int d = 0; d < 32; ++d) VsT[d][lane] = f2bf(vr[d]);   // transpose scatter
    __syncthreads();
    v8f s0, s1;
    {
      BFrag b0, b1;
      int col = lane & 15, kb2 = (lane & 16) ? 16 : 0;
      b0.w[0] = *(const v8us*)&Ks[col][kb2];
      b0.w[1] = *(const v8us*)&Ks[col][kb2 + 8];
      b1.w[0] = *(const v8us*)&Ks[col + 16][kb2];
      b1.w[1] = *(const v8us*)&Ks[col + 16][kb2 + 8];
      s0 = wmma_bf16(qa.v, b0.v, v8zero());
      s1 = wmma_bf16(qa.v, b1.v, v8zero());
    }
    float p0[8], p1[8];
#pragma unroll
    for (int i = 0; i < 8; ++i) {
      float mx = fmaxf(s0[i], s1[i]);
      mx = fmaxf(mx, __shfl_xor(mx, 1, 32));
      mx = fmaxf(mx, __shfl_xor(mx, 2, 32));
      mx = fmaxf(mx, __shfl_xor(mx, 4, 32));
      mx = fmaxf(mx, __shfl_xor(mx, 8, 32));
      float nm = fmaxf(rm[i], mx);
      float corr = __expf(rm[i] - nm);
      p0[i] = __expf(s0[i] - nm);
      p1[i] = __expf(s1[i] - nm);
      float ps = p0[i] + p1[i];
      ps += __shfl_xor(ps, 1, 32);
      ps += __shfl_xor(ps, 2, 32);
      ps += __shfl_xor(ps, 4, 32);
      ps += __shfl_xor(ps, 8, 32);
      rs[i] = rs[i] * corr + ps;
      rm[i] = nm;
      o0[i] *= corr;
      o1[i] *= corr;
    }
    __syncthreads();
    {
      int colb = lane & 15, rb = (lane & 16) ? 8 : 0;
#pragma unroll
      for (int i = 0; i < 8; ++i) {
        Ps[rb + i][colb] = f2bf(p0[i]);
        Ps[rb + i][colb + 16] = f2bf(p1[i]);
      }
    }
    __syncthreads();
    {
      int row2 = lane & 15, kb = (lane & 16) ? 16 : 0;
      int col = lane & 15, kbv = (lane & 16) ? 16 : 0;
      BFrag pf, v0f, v1f;
      pf.w[0] = *(const v8us*)&Ps[row2][kb];
      pf.w[1] = *(const v8us*)&Ps[row2][kb + 8];
      v0f.w[0] = *(const v8us*)&VsT[col][kbv];
      v0f.w[1] = *(const v8us*)&VsT[col][kbv + 8];
      v1f.w[0] = *(const v8us*)&VsT[col + 16][kbv];
      v1f.w[1] = *(const v8us*)&VsT[col + 16][kbv + 8];
      o0 = wmma_bf16(pf.v, v0f.v, o0);
      o1 = wmma_bf16(pf.v, v1f.v, o1);
    }
    __syncthreads();
  }
  int col = lane & 15, rb = (lane & 16) ? 8 : 0;
#pragma unroll
  for (int i = 0; i < 8; ++i) {
    int l = qt * 16 + rb + i;
    float inv = 1.f / rs[i];
    size_t base = ((size_t)(batch * LL + l)) * CC + fc * SCD;
    attm[base + col]      = f2bf(o0[i] * inv);
    attm[base + 16 + col] = f2bf(o1[i] * inv);
  }
}

// ---------------- 6a. agg base init ----------------
__global__ void agg_init_k(const float* __restrict__ aq, const float* __restrict__ av,
                           float* __restrict__ agg) {
  int idx = blockIdx.x * blockDim.x + threadIdx.x;   // 2048*65
  if (idx >= 2048 * 65) return;
  int j = idx % 65, row = idx / 65;
  int m = row >> 6, s = row & 63;
  int batch = m >> 3, fc = m & 7;
  float v;
  if (j < 32)       v = aq[((size_t)(batch * SS + s)) * CC + fc * SCD + j];
  else if (j < 64)  v = av[((size_t)(batch * SS + s)) * CC + fc * SCD + (j - 32)];
  else              v = 1.f;
  agg[idx] = v;
}

// ---------------- 6b. top-1 routing + scatter ----------------
__global__ void route_scatter_k(const float* __restrict__ dqf, const float* __restrict__ dvf,
                                const float* __restrict__ aq, const float* __restrict__ da,
                                const float* __restrict__ db, float* __restrict__ agg) {
  __shared__ float an[64][32];
  int m = blockIdx.x;
  int batch = m >> 3, fc = m & 7;
  int tid = threadIdx.x;                 // 128
  if (tid < 64) {
    const float* ar = aq + ((size_t)(batch * SS + tid)) * CC + fc * SCD;
    float ss = 0.f, t[32];
#pragma unroll
    for (int k = 0; k < 32; ++k) { t[k] = ar[k]; ss += t[k] * t[k]; }
    float inv = 1.f / fmaxf(sqrtf(ss), 1e-12f);
#pragma unroll
    for (int k = 0; k < 32; ++k) an[tid][k] = t[k] * inv;
  }
  __syncthreads();
  float Aa = da[0], Bb = db[0];
  int l = blockIdx.y * blockDim.x + tid;
  if (l < LL) {
    const float* qr = dqf + ((size_t)(batch * LL + l)) * CC + fc * SCD;
    float q[32], ss = 0.f;
#pragma unroll
    for (int k = 0; k < 32; ++k) { q[k] = qr[k]; ss += q[k] * q[k]; }
    float inv = 1.f / fmaxf(sqrtf(ss), 1e-12f);
    float best = -1e30f; int bi = 0;
    for (int s = 0; s < SS; ++s) {
      float d = 0.f;
#pragma unroll
      for (int k = 0; k < 32; ++k) d += q[k] * an[s][k];
      float z = Aa * (d * inv) + Bb;
      if (z > best) { best = z; bi = s; }
    }
    float val = 1.f / (1.f + __expf(-best));
    const float* vr = dvf + ((size_t)(batch * LL + l)) * CC + fc * SCD;
    float* dst = agg + ((size_t)(m * SS + bi)) * 65;
#pragma unroll
    for (int k = 0; k < 32; ++k) atomicAdd(&dst[k], val * qr[k]);
#pragma unroll
    for (int k = 0; k < 32; ++k) atomicAdd(&dst[32 + k], val * vr[k]);
    atomicAdd(&dst[64], val);
  }
}

// ---------------- 7. finalize aggregate ----------------
__global__ void agg_final_k(const float* __restrict__ agg, float* __restrict__ apn,
                            float* __restrict__ avn) {
  int row = blockIdx.x * blockDim.x + threadIdx.x;   // 2048
  if (row >= 2048) return;
  int m = row >> 6, s = row & 63;
  const float* src = agg + ((size_t)((m ^ 16) * SS + s)) * 65;  // half-swap
  float w = src[64];
  float iw = 1.f / w;
  float q[32], ss = 0.f;
#pragma unroll
  for (int k = 0; k < 32; ++k) { q[k] = src[k] * iw; ss += q[k] * q[k]; }
  float inv = 1.f / fmaxf(sqrtf(ss), 1e-12f);
#pragma unroll
  for (int k = 0; k < 32; ++k) {
    apn[(size_t)row * 32 + k] = q[k] * inv;
    avn[(size_t)row * 32 + k] = src[32 + k] * iw;
  }
}

// ---------------- 8. full-res routing + dispatch -> bf16 coc ----------------
__global__ void route_dispatch_k(const float* __restrict__ xp, const float* __restrict__ apn,
                                 const float* __restrict__ avn, const float* __restrict__ al,
                                 const float* __restrict__ be,
                                 unsigned short* __restrict__ coc) {
  __shared__ float sap[64][32];
  __shared__ float sav[64][32];
  int m = blockIdx.x;
  int batch = m >> 3, fc = m & 7;
  int tid = threadIdx.x;                 // 256
  for (int i = tid; i < 64 * 32; i += 256) {
    sap[i >> 5][i & 31] = apn[(size_t)m * 2048 + i];
    sav[i >> 5][i & 31] = avn[(size_t)m * 2048 + i];
  }
  __syncthreads();
  float Aa = al[0], Bb = be[0];
  int pix = blockIdx.y * 256 + tid;      // grid.y = 100
  const float* xr = xp + ((size_t)(batch * PIX + pix)) * CC + fc * SCD;
  float q[32], ss = 0.f;
#pragma unroll
  for (int k = 0; k < 32; ++k) { q[k] = xr[k]; ss += q[k] * q[k]; }
  float inv = 1.f / fmaxf(sqrtf(ss), 1e-12f);
  float best = -1e30f; int bi = 0;
  for (int s = 0; s < SS; ++s) {
    float d = 0.f;
#pragma unroll
    for (int k = 0; k < 32; ++k) d += q[k] * sap[s][k];
    float z = Aa * (d * inv) + Bb;
    if (z > best) { best = z; bi = s; }
  }
  float val = 1.f / (1.f + __expf(-best));
  unsigned short* dst = coc + ((size_t)(batch * PIX + pix)) * CC + fc * SCD;
#pragma unroll
  for (int k = 0; k < 32; ++k) dst[k] = f2bf(val * sav[bi][k]);
}

// ---------------- 9. bilinear 4x upsample -> out[:,0:256] NCHW ----------------
__global__ void resize_k(const float* __restrict__ msg, float* __restrict__ out) {
  size_t idx = (size_t)blockIdx.x * blockDim.x + threadIdx.x;  // 4*256*160*160
  int w = idx % WW;
  int h = (idx / WW) % HH;
  int c = (idx / PIX) % CC;
  int n = idx / ((size_t)CC * PIX);
  float sy = (h + 0.5f) * 0.25f - 0.5f;
  float sx = (w + 0.5f) * 0.25f - 0.5f;
  int y0 = (int)floorf(sy), x0 = (int)floorf(sx);
  float fy = sy - (float)y0, fx = sx - (float)x0;
  int y0c = min(max(y0, 0), SH - 1), y1c = min(max(y0 + 1, 0), SH - 1);
  int x0c = min(max(x0, 0), SW - 1), x1c = min(max(x0 + 1, 0), SW - 1);
  const float* base = msg + (size_t)n * LL * CC + c;
  float g00 = base[(size_t)(y0c * SW + x0c) * CC];
  float g01 = base[(size_t)(y0c * SW + x1c) * CC];
  float g10 = base[(size_t)(y1c * SW + x0c) * CC];
  float g11 = base[(size_t)(y1c * SW + x1c) * CC];
  float v = (1.f - fy) * ((1.f - fx) * g00 + fx * g01) + fy * ((1.f - fx) * g10 + fx * g11);
  out[((size_t)(n * 512 + c)) * PIX + h * WW + w] = v;
}

// ------------------------------ launcher ------------------------------
extern "C" void kernel_launch(void* const* d_in, const int* in_sizes, int n_in,
                              void* d_out, int out_size, void* d_ws, size_t ws_size,
                              hipStream_t stream) {
  const float* x        = (const float*)d_in[0];
  const float* w_down   = (const float*)d_in[1];
  const float* b_down   = (const float*)d_in[2];
  const float* ln_w     = (const float*)d_in[3];
  const float* ln_b     = (const float*)d_in[4];
  const float* w_qk     = (const float*)d_in[5];
  const float* b_qk     = (const float*)d_in[6];
  const float* w_v      = (const float*)d_in[7];
  const float* b_v      = (const float*)d_in[8];
  const float* w_point  = (const float*)d_in[9];
  const float* b_point  = (const float*)d_in[10];
  const float* d_alpha  = (const float*)d_in[11];
  const float* d_beta   = (const float*)d_in[12];
  const float* alpha    = (const float*)d_in[13];
  const float* beta     = (const float*)d_in[14];
  const float* w_m0     = (const float*)d_in[15];
  const float* b_m0     = (const float*)d_in[16];
  const float* w_m1     = (const float*)d_in[17];
  const float* b_m1     = (const float*)d_in[18];
  float* out = (float*)d_out;

  float* p = (float*)d_ws;
  float* dq_n = p; p += (size_t)6400 * 256;
  float* dv_n = p; p += (size_t)6400 * 256;
  float* dq_f = p; p += (size_t)6400 * 256;
  float* dv_f = p; p += (size_t)6400 * 256;
  float* msgs = p; p += (size_t)6400 * 256;
  float* aq   = p; p += (size_t)4 * 64 * 256;
  float* av   = p; p += (size_t)4 * 64 * 256;
  float* agg  = p; p += (size_t)2048 * 65;
  float* apn  = p; p += (size_t)2048 * 32;
  float* avn  = p; p += (size_t)2048 * 32;
  float* xp   = p; p += (size_t)102400 * 256;
  unsigned short* q = (unsigned short*)p;
  unsigned short* dq_b = q; q += (size_t)6400 * 256;
  unsigned short* dv_b = q; q += (size_t)6400 * 256;
  unsigned short* at_b = q; q += (size_t)6400 * 256;
  unsigned short* xb   = q; q += (size_t)102400 * 256;
  unsigned short* cocb = q; q += (size_t)102400 * 256;
  unsigned short* wtqk = q; q += (size_t)256 * 256;
  unsigned short* wtv  = q; q += (size_t)256 * 256;
  unsigned short* wtp  = q; q += (size_t)256 * 256;
  unsigned short* wtm0 = q; q += (size_t)256 * 256;
  unsigned short* wtm1 = q; q += (size_t)256 * 256;

  // 1. downsample (conv + maxpool)
  downsample_k<<<6400, 256, 0, stream>>>(x, w_down, b_down, dq_n, dv_n);
  // 2. layernorm -> bf16 GEMM operands
  ln_k<<<6400, 256, 0, stream>>>(dq_n, ln_w, ln_b, dq_b);
  ln_k<<<6400, 256, 0, stream>>>(dv_n, ln_w, ln_b, dv_b);
  // 2b. weights transposed -> bf16 ; x NCHW -> NHWC bf16
  wtrans_k<<<256, 256, 0, stream>>>(w_qk, wtqk);
  wtrans_k<<<256, 256, 0, stream>>>(w_v, wtv);
  wtrans_k<<<256, 256, 0, stream>>>(w_point, wtp);
  wtrans_k<<<256, 256, 0, stream>>>(w_m0, wtm0);
  wtrans_k<<<256, 256, 0, stream>>>(w_m1, wtm1);
  xconv_k<<<102400, 256, 0, stream>>>(x, xb);
  // 3. qk / v projections (TDM + WMMA GEMM)
  gemm256_k<<<dim3(100, 8), 128, 0, stream>>>(dq_b, wtqk, b_qk, dq_f, -1);
  gemm256_k<<<dim3(100, 8), 128, 0, stream>>>(dv_b, wtv, b_v, dv_f, -1);
  // 4. anchors
  anchor_pool_k<<<256, 256, 0, stream>>>(dq_f, aq);
  anchor_pool_k<<<256, 256, 0, stream>>>(dv_f, av);
  // 5. bidirectional flash attention (WMMA)
  attention_k<<<dim3(100, 32), 32, 0, stream>>>(dq_f, dv_f, at_b);
  // 6. message projection m0 + 4x bilinear upsample -> out[:, 0:256]
  gemm256_k<<<dim3(100, 8), 128, 0, stream>>>(at_b, wtm0, b_m0, msgs, -1);
  resize_k<<<102400, 256, 0, stream>>>(msgs, out);
  // 7. full-res point features (TDM + WMMA GEMM)
  gemm256_k<<<dim3(1600, 8), 128, 0, stream>>>(xb, wtp, b_point, xp, -1);
  // 8. routing + scatter aggregation
  agg_init_k<<<(2048 * 65 + 255) / 256, 256, 0, stream>>>(aq, av, agg);
  route_scatter_k<<<dim3(32, 13), 128, 0, stream>>>(dq_f, dv_f, aq, d_alpha, d_beta, agg);
  agg_final_k<<<8, 256, 0, stream>>>(agg, apn, avn);
  // 9. full-res routing + dispatch -> bf16
  route_dispatch_k<<<dim3(32, 100), 256, 0, stream>>>(xp, apn, avn, alpha, beta, cocb);
  // 10. m1 projection straight into out[:, 256:512] NCHW (TDM + WMMA GEMM)
  gemm256_k<<<dim3(1600, 8), 128, 0, stream>>>(cocb, wtm1, b_m1, out, 256);
}
